// BaseController_73684458930500
// MI455X (gfx1250) — matrix operand order repaired and checked by reading, not compiled, not run on hardware
//
#include <hip/hip_runtime.h>
#include <hip/hip_bf16.h>
#include <stdint.h>

// Problem constants (match reference module; k is fixed at trace time)
#define H      1024
#define E      20
#define MLP    1000
#define NW     4096
#define NC     81920
#define KSEL   1638
#define STRIDE 1024   // padded column stride for A/B tables
#define GM     64     // GEMM block tile M
#define GN     64     // GEMM block tile N

typedef __attribute__((ext_vector_type(16))) __bf16 v16bf;
typedef __attribute__((ext_vector_type(8)))  float  v8f;
typedef unsigned int u32x4 __attribute__((ext_vector_type(4)));
typedef int          i32x4 __attribute__((ext_vector_type(4)));
typedef int          i32x8 __attribute__((ext_vector_type(8)));

__device__ __forceinline__ unsigned short f2bf(float f) {
    union { float f; uint32_t u; } c; c.f = f;
    uint32_t u = c.u;
    uint32_t r = u + 0x7FFFu + ((u >> 16) & 1u);   // round-to-nearest-even
    if ((u & 0x7F800000u) == 0x7F800000u) r = u;   // inf/nan passthrough
    return (unsigned short)(r >> 16);
}

// ---------------------------------------------------------------------------
// TDM: issue a 2D tensor_load_to_lds for a tile of 2-byte elements.
//   D# group0: count=1, lds_addr, global_addr[56:0], type=2  (ISA 8.3)
//   D# group1: mask=0, data_size=1(2B), tensor_dim0/1, tile_dim0/1,
//              tensor_dim0_stride (elements)                 (ISA 8.4)
// Groups 2/3 zero (2D tensor). Tracked by TENSORcnt.
// ---------------------------------------------------------------------------
__device__ __forceinline__ void tdm_load_2d(unsigned lds_off, const void* gaddr,
                                            unsigned tile_d0, unsigned tile_d1,
                                            unsigned tensor_d0, unsigned tensor_d1,
                                            unsigned stride0_elems) {
    unsigned long long ga = (unsigned long long)(uintptr_t)gaddr;
    u32x4 g0;
    g0.x = 1u;                                                  // count=1
    g0.y = lds_off;                                             // lds_addr (bytes)
    g0.z = (unsigned)(ga & 0xFFFFFFFFu);                        // global_addr[31:0]
    g0.w = (unsigned)((ga >> 32) & 0x01FFFFFFu) | (2u << 30);   // [56:32] | type=2
    i32x8 g1;
    g1[0] = (int)(1u << 16);                                    // mask=0, data_size=1 (2B)
    g1[1] = (int)((tensor_d0 & 0xFFFFu) << 16);                 // tensor_dim0[15:0]
    g1[2] = (int)((tensor_d0 >> 16) | ((tensor_d1 & 0xFFFFu) << 16));
    g1[3] = (int)((tensor_d1 >> 16) | (tile_d0 << 16));         // tile_dim0
    g1[4] = (int)(tile_d1 & 0xFFFFu);                           // tile_dim1, tile_dim2=0
    g1[5] = (int)stride0_elems;                                 // tensor_dim0_stride[31:0]
    g1[6] = 0;
    g1[7] = 0;
    i32x4 z4 = {0, 0, 0, 0};
#if __clang_major__ >= 23
    i32x8 z8 = {0, 0, 0, 0, 0, 0, 0, 0};
    __builtin_amdgcn_tensor_load_to_lds(g0, g1, z4, z4, z8, 0);
#else
    __builtin_amdgcn_tensor_load_to_lds(g0, g1, z4, z4, 0);
#endif
}

// ---------------------------------------------------------------------------
// Kernel P: pack doc -> bf16 row-major; pack W1[0:2048] -> bf16 transposed
// WtB[z][n][k] = bf16(W1[(z*H + k)*MLP + n]), zero for n >= MLP.
// ---------------------------------------------------------------------------
__global__ void pack_kernel(const float* __restrict__ doc,
                            const float* __restrict__ W1,
                            unsigned short* __restrict__ docbf,
                            unsigned short* __restrict__ WtB) {
    const size_t DOCN = (size_t)NW * H;
    size_t i = (size_t)blockIdx.x * 256 + threadIdx.x;
    if (i < DOCN) {
        docbf[i] = f2bf(doc[i]);
    } else {
        size_t j = i - DOCN;
        int z   = (int)(j >> 20);
        int rem = (int)(j & 0xFFFFFu);
        int k   = rem >> 10;
        int nn  = rem & 1023;
        float v = (nn < MLP) ? W1[(size_t)(z * H + k) * MLP + nn] : 0.0f;
        WtB[(((size_t)z << 20) | ((size_t)nn << 10)) + (size_t)k] = f2bf(v);
    }
}

// ---------------------------------------------------------------------------
// Kernel G: A = doc @ W1[0:H], B = doc @ W1[H:2H] with TDM-staged LDS tiles
// and v_wmma_f32_16x16x32_bf16. Block tile 64x64: 8 waves = 4(M) x 2(N-half),
// each wave owns TWO 16x16 accumulators sharing one A fragment.
// Double-buffered 64x32 bf16 doc tile + 64x32 bf16 W^T tile in LDS.
// A/B fragments use contiguous k = 16*(lane/16) + s for both operands (wave32).
// ---------------------------------------------------------------------------
__global__ void __launch_bounds__(256) gemm_kernel(const unsigned short* __restrict__ docbf,
                                                   const unsigned short* __restrict__ WtB,
                                                   float* __restrict__ A, float* __restrict__ B) {
    __shared__ __align__(16) unsigned short sA[2][GM * 32];   // 4 KB per buffer
    __shared__ __align__(16) unsigned short sB[2][GN * 32];   // 4 KB per buffer

    const int wave = threadIdx.x >> 5;
    const int lane = threadIdx.x & 31;
    const int z  = blockIdx.z;
    const int m0 = blockIdx.y * GM;
    const int n0 = blockIdx.x * GN;
    const unsigned short* wbase = WtB + ((size_t)z << 20);
    float* out = z ? B : A;

    // LDS aperture: generic shared pointer's low 32 bits are the LDS byte offset
    const unsigned offA0 = (unsigned)(uintptr_t)&sA[0][0];
    const unsigned offA1 = (unsigned)(uintptr_t)&sA[1][0];
    const unsigned offB0 = (unsigned)(uintptr_t)&sB[0][0];
    const unsigned offB1 = (unsigned)(uintptr_t)&sB[1][0];

    if (wave == 0) {
        tdm_load_2d(offA0, docbf + (size_t)m0 * 1024, 32, GM, 1024, NW, 1024);
        tdm_load_2d(offB0, wbase + (size_t)n0 * 1024, 32, GN, 1024, 1024, 1024);
        __builtin_amdgcn_s_wait_tensorcnt(0);
    }
    __syncthreads();

    const int mt  = wave >> 1;                 // 0..3  (M sub-tile)
    const int nh  = (wave & 1) * 32;           // N half: 0 or 32
    const int ml  = mt * 16 + (lane & 15);
    const int nl0 = nh + (lane & 15);          // first N tile row in W^T
    const int nl1 = nl0 + 16;                  // second N tile row in W^T
    const int khalfA = (lane >> 4) * 16;       // A: K half {0-15} vs {16-31}
    const int kbaseB = (lane >> 4) * 16;       // B: K half {0-15} vs {16-31}

    v8f c0 = {}, c1 = {};
    for (int ks = 0; ks < 32; ++ks) {
        const int cur = ks & 1;
        const bool more = (ks + 1) < 32;
        if (more && wave == 0) {               // prefetch next K-slab into other buffer
            const int k0 = (ks + 1) * 32;
            tdm_load_2d(cur ? offA0 : offA1, docbf + (size_t)m0 * 1024 + k0, 32, GM, 1024, NW, 1024);
            tdm_load_2d(cur ? offB0 : offB1, wbase + (size_t)n0 * 1024 + k0, 32, GN, 1024, 1024, 1024);
        }
        union { uint4 u[2]; v16bf v; } af, b0, b1;
        const unsigned short* pa  = &sA[cur][ml * 32];
        const unsigned short* pb0 = &sB[cur][nl0 * 32];
        const unsigned short* pb1 = &sB[cur][nl1 * 32];
        af.u[0] = *(const uint4*)(pa + khalfA);          // ds_load_b128 x6
        af.u[1] = *(const uint4*)(pa + khalfA + 8);
        b0.u[0] = *(const uint4*)(pb0 + kbaseB);
        b0.u[1] = *(const uint4*)(pb0 + kbaseB + 8);
        b1.u[0] = *(const uint4*)(pb1 + kbaseB);
        b1.u[1] = *(const uint4*)(pb1 + kbaseB + 8);
        c0 = __builtin_amdgcn_wmma_f32_16x16x32_bf16(
                 false, af.v, false, b0.v, (short)0, c0, false, false);
        c1 = __builtin_amdgcn_wmma_f32_16x16x32_bf16(
                 false, af.v, false, b1.v, (short)0, c1, false, false);
        if (more && wave == 0) __builtin_amdgcn_s_wait_tensorcnt(0);
        __syncthreads();                       // publish next buffer / protect cur
    }

    const int mb = m0 + mt * 16 + 8 * (lane >> 4);
    const int nc0 = n0 + nl0;
    const int nc1 = n0 + nl1;
#pragma unroll
    for (int r = 0; r < 8; ++r) {
        out[(size_t)(mb + r) * STRIDE + nc0] = c0[r];
        out[(size_t)(mb + r) * STRIDE + nc1] = c1[r];
    }
}

// ---------------------------------------------------------------------------
// Kernel W: per-width tables (only 20 distinct widths)
// ---------------------------------------------------------------------------
__global__ void width_kernel(const float* __restrict__ we,  const float* __restrict__ wpe,
                             const float* __restrict__ W1,  const float* __restrict__ b1,
                             const float* __restrict__ b2,  const float* __restrict__ Wp1,
                             const float* __restrict__ bp1, const float* __restrict__ wp2,
                             const float* __restrict__ bp2,
                             float* __restrict__ Cb, float* __restrict__ wscore) {
    const int w = blockIdx.x;
    __shared__ float red[256];
    float wrow[E], prow[E];
#pragma unroll
    for (int e = 0; e < E; ++e) { wrow[e] = we[w * E + e]; prow[e] = wpe[w * E + e]; }

    float partial = 0.0f;
    for (int j = threadIdx.x; j < MLP; j += 256) {
        float cb = b1[j];
        float hp = bp1[j];
#pragma unroll
        for (int e = 0; e < E; ++e) {
            cb = fmaf(wrow[e], W1[(size_t)(2 * H + e) * MLP + j], cb);
            hp = fmaf(prow[e], Wp1[(size_t)e * MLP + j], hp);
        }
        Cb[(size_t)w * MLP + j] = cb;
        partial += fmaxf(hp, 0.0f) * wp2[j];
    }
    red[threadIdx.x] = partial;
    __syncthreads();
    for (int s = 128; s > 0; s >>= 1) {
        if (threadIdx.x < (unsigned)s) red[threadIdx.x] += red[threadIdx.x + s];
        __syncthreads();
    }
    if (threadIdx.x == 0) wscore[w] = red[0] + b2[0] + bp2[0];
}

// ---------------------------------------------------------------------------
// Kernel S: one wave per candidate; A/B/Cb rows are L2-resident (33 MB << 192 MB)
// ---------------------------------------------------------------------------
__global__ void score_kernel(const int* __restrict__ starts, const int* __restrict__ widths,
                             const float* __restrict__ A, const float* __restrict__ B,
                             const float* __restrict__ Cb, const float* __restrict__ w2,
                             const float* __restrict__ wscore, float* __restrict__ logits) {
    const int wave = threadIdx.x >> 5, lane = threadIdx.x & 31;
    const int n = blockIdx.x * 8 + wave;
    const int s = starts[n];
    const int w = widths[n];
    const int e = s + w;
    const float* ar = A + (size_t)s * STRIDE;
    const float* br = B + (size_t)e * STRIDE;
    const float* cr = Cb + (size_t)w * MLP;

    float acc = 0.0f;
#pragma unroll
    for (int it = 0; it < 7; ++it) {
        int j = it * 128 + lane * 4;
        float4 a  = *(const float4*)(ar + j);
        float4 b  = *(const float4*)(br + j);
        float4 c  = *(const float4*)(cr + j);
        float4 ww = *(const float4*)(w2 + j);
        acc = fmaf(fmaxf(a.x + b.x + c.x, 0.0f), ww.x, acc);
        acc = fmaf(fmaxf(a.y + b.y + c.y, 0.0f), ww.y, acc);
        acc = fmaf(fmaxf(a.z + b.z + c.z, 0.0f), ww.z, acc);
        acc = fmaf(fmaxf(a.w + b.w + c.w, 0.0f), ww.w, acc);
    }
    {
        int j = 896 + lane * 4;
#pragma unroll
        for (int t = 0; t < 4; ++t)
            if (j + t < MLP)
                acc = fmaf(fmaxf(ar[j + t] + br[j + t] + cr[j + t], 0.0f), w2[j + t], acc);
    }
    for (int off = 16; off > 0; off >>= 1)
        acc += __shfl_xor(acc, off, 32);
    if (lane == 0) logits[n] = acc + wscore[w];
}

// ---------------------------------------------------------------------------
// Top-k pipeline: histogram prefilter + exact rank among survivors.
// Monotone key: m = (u & sign) ? ~u : (u | sign); bin = m >> 20 (4096 bins).
// ---------------------------------------------------------------------------
__device__ __forceinline__ unsigned fkey(float f) {
    union { float f; unsigned u; } c; c.f = f;
    return (c.u & 0x80000000u) ? ~c.u : (c.u | 0x80000000u);
}

__global__ void zero_hist_kernel(unsigned* __restrict__ hist) {
    hist[blockIdx.x * 256 + threadIdx.x] = 0u;
}

__global__ void hist_kernel(const float* __restrict__ logits, unsigned* __restrict__ hist) {
    __shared__ unsigned h[4096];
    for (int t = threadIdx.x; t < 4096; t += 256) h[t] = 0u;
    __syncthreads();
    const int gid = blockIdx.x * 256 + threadIdx.x;     // 80 blocks
    for (int i = gid; i < NC; i += 80 * 256)
        atomicAdd(&h[fkey(logits[i]) >> 20], 1u);
    __syncthreads();
    for (int t = threadIdx.x; t < 4096; t += 256)
        if (h[t]) atomicAdd(&hist[t], h[t]);
}

__global__ void thresh_kernel(const unsigned* __restrict__ hist, unsigned* __restrict__ meta) {
    if (threadIdx.x == 0) {
        unsigned acc = 0; int b = 4095;
        for (; b > 0; --b) { acc += hist[b]; if (acc >= KSEL) break; }
        if (acc < KSEL) b = 0;
        meta[0] = (unsigned)b;    // threshold bin
        meta[1] = 0u;             // survivor counter (reset every launch/replay)
    }
}

__global__ void compact_kernel(const float* __restrict__ logits,
                               const int* __restrict__ starts, const int* __restrict__ widths,
                               unsigned* __restrict__ meta,
                               float* __restrict__ svLogit, int* __restrict__ svIdx,
                               int* __restrict__ svS, int* __restrict__ svE) {
    const int i = blockIdx.x * 256 + threadIdx.x;
    const float x = logits[i];
    if ((fkey(x) >> 20) >= meta[0]) {
        unsigned slot = atomicAdd(&meta[1], 1u);   // order-free: final output is rank-placed
        svLogit[slot] = x;
        svIdx[slot]   = i;
        int s = starts[i], w = widths[i];
        svS[slot] = s;
        svE[slot] = s + w;
    }
}

__global__ void rank_kernel(const unsigned* __restrict__ meta,
                            const float* __restrict__ svLogit, const int* __restrict__ svIdx,
                            const int* __restrict__ svS, const int* __restrict__ svE,
                            int* __restrict__ selS, int* __restrict__ selE,
                            float* __restrict__ selScore, float* __restrict__ selKey) {
    __shared__ float tl[1024];
    __shared__ int   ti[1024];
    const int S = (int)meta[1];
    if (blockIdx.x * 256 >= S) return;                 // uniform per block
    const int t0 = blockIdx.x * 256 + threadIdx.x;
    const bool act = t0 < S;
    float x = 0.0f; int idx = 0x7FFFFFFF;
    if (act) { x = svLogit[t0]; idx = svIdx[t0]; }
    int cnt = 0;
    for (int j0 = 0; j0 < S; j0 += 1024) {
        __syncthreads();
        for (int t = threadIdx.x; t < 1024; t += 256) {
            int j = j0 + t;
            tl[t] = (j < S) ? svLogit[j] : -3.0e38f;
            ti[t] = (j < S) ? svIdx[j] : 0x7FFFFFFF;
        }
        __syncthreads();
        const int lim = min(1024, S - j0);
        if (act)
            for (int t = 0; t < lim; ++t) {
                float y = tl[t];
                cnt += (y > x) || (y == x && ti[t] < idx);   // jax top_k tie: lower index wins
            }
    }
    if (act && cnt < KSEL) {                            // global rank == rank among survivors
        selS[cnt]     = svS[t0];
        selE[cnt]     = svE[t0];
        selScore[cnt] = x;
        selKey[cnt]   = (float)svS[t0] + 1e-5f * (float)svE[t0];  // reference's f32 key
    }
}

// ---------------------------------------------------------------------------
// Kernel T2: stable position-sort of the KSEL survivors by rank-counting
// ---------------------------------------------------------------------------
__global__ void sort_kernel(const int* __restrict__ selS, const int* __restrict__ selE,
                            const float* __restrict__ selScore, const float* __restrict__ selKey,
                            float* __restrict__ out) {
    __shared__ float keys[KSEL];
    const int r = blockIdx.x * 256 + threadIdx.x;
    for (int t = threadIdx.x; t < KSEL; t += 256) keys[t] = selKey[t];
    __syncthreads();
    if (r < KSEL) {
        float kr = keys[r];
        int pr = 0;
        for (int q = 0; q < KSEL; ++q) {
            float kq = keys[q];
            pr += (kq < kr) || (kq == kr && q < r);    // stable argsort
        }
        out[pr]            = (float)selS[r];
        out[KSEL + pr]     = (float)selE[r];
        out[2 * KSEL + pr] = selScore[r];
    }
}

// ---------------------------------------------------------------------------
extern "C" void kernel_launch(void* const* d_in, const int* in_sizes, int n_in,
                              void* d_out, int out_size, void* d_ws, size_t ws_size,
                              hipStream_t stream) {
    const float* doc    = (const float*)d_in[0];
    const int*   starts = (const int*)  d_in[1];
    const int*   widths = (const int*)  d_in[2];
    const float* we     = (const float*)d_in[3];
    const float* wpe    = (const float*)d_in[4];
    const float* W1     = (const float*)d_in[5];
    const float* b1     = (const float*)d_in[6];
    const float* w2     = (const float*)d_in[7];
    const float* b2     = (const float*)d_in[8];
    const float* Wp1    = (const float*)d_in[9];
    const float* bp1    = (const float*)d_in[10];
    const float* wp2    = (const float*)d_in[11];
    const float* bp2    = (const float*)d_in[12];
    (void)n_in; (void)in_sizes; (void)ws_size;

    char* ws = (char*)d_ws;
    float*          Abuf    = (float*)(ws);                     // 16 MB
    float*          Bbuf    = (float*)(ws + 16777216);          // 16 MB
    unsigned short* docbf   = (unsigned short*)(ws + 33554432); // 8 MB
    unsigned short* WtB     = (unsigned short*)(ws + 41943040); // 4 MB
    float*          Cb      = (float*)(ws + 46137344);          // 80 KB
    float*          wsc     = (float*)(ws + 46219264);
    float*          logits  = (float*)(ws + 46219520);          // 320 KB
    unsigned*       hist    = (unsigned*)(ws + 46547200);       // 16 KB
    unsigned*       meta    = (unsigned*)(ws + 46563584);
    float*          svLogit = (float*)(ws + 46563840);          // 320 KB
    int*            svIdx   = (int*)  (ws + 46891520);          // 320 KB
    int*            svS     = (int*)  (ws + 47219200);          // 320 KB
    int*            svE     = (int*)  (ws + 47546880);          // 320 KB
    int*            selS    = (int*)  (ws + 47874560);
    int*            selE    = (int*)  (ws + 47882752);
    float*          selSc   = (float*)(ws + 47890944);
    float*          selKey  = (float*)(ws + 47899136);

    // 1) pack doc -> bf16, W1[0:2048] -> transposed bf16 (zero-padded cols)
    {
        const size_t total = (size_t)NW * H + 2u * 1024u * 1024u;
        pack_kernel<<<dim3((unsigned)(total / 256)), dim3(256), 0, stream>>>(doc, W1, docbf, WtB);
    }
    // 2) width tables
    width_kernel<<<dim3(20), dim3(256), 0, stream>>>(we, wpe, W1, b1, b2, Wp1, bp1, wp2, bp2, Cb, wsc);
    // 3) WMMA GEMMs with TDM double-buffered LDS staging (64x64 block tiles)
    gemm_kernel<<<dim3(1024 / GN, NW / GM, 2), dim3(256), 0, stream>>>(docbf, WtB, Abuf, Bbuf);
    // 4) per-candidate scoring
    score_kernel<<<dim3(NC / 8), dim3(256), 0, stream>>>(starts, widths, Abuf, Bbuf, Cb, w2, wsc, logits);
    // 5) top-k: histogram prefilter -> threshold bin -> compact -> exact rank
    zero_hist_kernel<<<dim3(16), dim3(256), 0, stream>>>(hist);
    hist_kernel<<<dim3(80), dim3(256), 0, stream>>>(logits, hist);
    thresh_kernel<<<dim3(1), dim3(256), 0, stream>>>(hist, meta);
    compact_kernel<<<dim3(NC / 256), dim3(256), 0, stream>>>(logits, starts, widths, meta,
                                                             svLogit, svIdx, svS, svE);
    rank_kernel<<<dim3(NC / 256), dim3(256), 0, stream>>>(meta, svLogit, svIdx, svS, svE,
                                                          selS, selE, selSc, selKey);
    // 6) stable position sort + emit (starts | ends | scores)
    sort_kernel<<<dim3((KSEL + 255) / 256), dim3(256), 0, stream>>>(selS, selE, selSc, selKey,
                                                                    (float*)d_out);
    (void)out_size;
}
